// ShapeEncoder_33165737459942
// MI455X (gfx1250) — matrix operand order repaired and checked by reading, not compiled, not run on hardware
//
#include <hip/hip_runtime.h>

typedef __attribute__((ext_vector_type(16))) _Float16 v16h;
typedef __attribute__((ext_vector_type(8)))  _Float16 v8h;
typedef __attribute__((ext_vector_type(8)))  float    v8f;

#define NNODES   65536
#define NGRAPHS  32
#define NEDGES   393216
#define BN_EPS   1e-5f
#define ENC_NEGINF 0x007FFFFFu
#define NPART    512

// ---- order-preserving float<->u32 encoding (max is order-independent -> deterministic) ----
static __device__ __forceinline__ unsigned enc_f32(float x) {
  unsigned u = __float_as_uint(x);
  return (u & 0x80000000u) ? ~u : (u | 0x80000000u);
}
static __device__ __forceinline__ float dec_f32(unsigned e) {
  unsigned b = (e & 0x80000000u) ? (e & 0x7FFFFFFFu) : ~e;
  return __uint_as_float(b);
}

// ---- small conversion kernels ----
__global__ void conv_f32_f16(const float* __restrict__ a, _Float16* __restrict__ b, size_t n) {
  size_t i = (size_t)blockIdx.x * blockDim.x + threadIdx.x;
  if (i < n) b[i] = (_Float16)a[i];
}

// Pack W [K,C] f32 into WMMA-B fragment order: Wp[colTile][Kpad/32][lane 0..31][j 0..15] f16.
// fragment element j of lane L in chunk c == W[c*32 + (L>>4)*8 + (j&7) + (j>>3)*16][colTile*16 + (L&15)]
__global__ void conv_w_pack(const float* __restrict__ W, _Float16* __restrict__ Wp,
                            int K, int C, int Kpad) {
  size_t i = (size_t)blockIdx.x * blockDim.x + threadIdx.x;
  if (i >= (size_t)Kpad * C) return;
  int nch  = Kpad >> 5;
  int j    = (int)(i & 15);
  int lane = (int)((i >> 4) & 31);
  int rest = (int)(i >> 9);            // t*nch + c
  int c = rest % nch, t = rest / nch;
  int k   = c * 32 + (lane >> 4) * 8 + (j & 7) + (j >> 3) * 16;
  int col = t * 16 + (lane & 15);
  Wp[i] = (k < K) ? (_Float16)W[(size_t)k * C + col] : (_Float16)0.0f;
}

__global__ void fill_u32(unsigned* __restrict__ p, size_t n, unsigned v) {
  size_t i = (size_t)blockIdx.x * blockDim.x + threadIdx.x;
  if (i < n) p[i] = v;
}

// ---- WMMA tile compute: two 16-row sub-tiles per wave share one B fragment ----
// As: fragment-ordered LDS, [2][K/32][32 lanes][16 halves]; Wp fragment-ordered in HBM.
template<int K, int C>
static __device__ __forceinline__ void wmma_tile_compute2(
    const _Float16* As, const _Float16* __restrict__ Wp,
    const float* __restrict__ bias, float* __restrict__ Y, int row0) {
  constexpr int NCH = K / 32;
  const int wave = threadIdx.x >> 5;
  const int lane = threadIdx.x & 31;
  const int colTile = blockIdx.y * 8 + wave;
  if (colTile * 16 >= C) return;
  const _Float16* wp = Wp + (size_t)colTile * NCH * 512;
  v8f acc0 = {}, acc1 = {};
  #pragma unroll
  for (int c = 0; c < NCH; ++c) {
    v16h b  = *(const v16h*)&wp[(c * 32 + lane) * 16];          // coalesced 32B/lane
    v16h a0 = *(const v16h*)&As[(c * 32 + lane) * 16];          // 2x ds_load_b128
    v16h a1 = *(const v16h*)&As[16 * K + (c * 32 + lane) * 16];
    acc0 = __builtin_amdgcn_wmma_f32_16x16x32_f16(false, a0, false, b, (short)0, acc0,
                                                  false, false);
    acc1 = __builtin_amdgcn_wmma_f32_16x16x32_f16(false, a1, false, b, (short)0, acc1,
                                                  false, false);
  }
  const int n = lane & 15, half = lane >> 4;
  const int gcol = colTile * 16 + n;
  const float bv = bias[gcol];
  float* y0 = Y + (size_t)(row0 + half * 8) * C + gcol;         // C constexpr -> imm offsets
  #pragma unroll
  for (int v = 0; v < 8; ++v) y0[(size_t)v * C] = fmaxf(acc0[v] + bv, 0.0f);
  float* y1 = Y + (size_t)(row0 + 16 + half * 8) * C + gcol;
  #pragma unroll
  for (int v = 0; v < 8; ++v) y1[(size_t)v * C] = fmaxf(acc1[v] + bv, 0.0f);
}

// Row-major (r,k) -> fragment slot: lane16 = ((kk>>3)&1)*16 + (r&15), j = (kk&7)|((kk&16)>>1).
// 8-aligned K groups map to 8 contiguous j -> vector LDS stores.

// ---- generic GEMM: Y[R,C] = relu(A[R,K]*W + bias); A packed f16 row-major; 32 rows/block ----
template<int K, int C>
__global__ __launch_bounds__(256)
void wmma_gemm_t(const _Float16* __restrict__ A, const _Float16* __restrict__ Wp,
                 const float* __restrict__ bias, float* __restrict__ Y) {
  __shared__ _Float16 As[32 * K] __attribute__((aligned(32)));
  const int row0 = blockIdx.x * 32;
  for (int g = threadIdx.x; g < 4 * K; g += 256) {       // 32*K/8 vector groups
    int r = g / (K / 8), kb = g % (K / 8);
    int k8 = kb * 8, c = k8 >> 5, kk = k8 & 31;
    int sub = r >> 4, lane16 = ((kk >> 3) & 1) * 16 + (r & 15);
    int jb = (kk & 16) >> 1;
    v8h av = *(const v8h*)&A[(size_t)(row0 + r) * K + k8];
    *(v8h*)&As[sub * 16 * K + (c * 32 + lane16) * 16 + jb] = av;
  }
  __syncthreads();
  wmma_tile_compute2<K, C>(As, Wp, bias, Y, row0);
}

// ---- fused EdgeConv layer-0 GEMM: gathers [x_dst, x_src - x_dst] straight into
// fragment-ordered LDS (no intermediate tile). Vectorized path when CIN % 8 == 0. ----
template<int K, int C, int CIN>
__global__ __launch_bounds__(256)
void wmma_edge_gemm_t(const _Float16* __restrict__ x, int ldx,
                      const int* __restrict__ src, const int* __restrict__ dst,
                      const _Float16* __restrict__ Wp, const float* __restrict__ bias,
                      float* __restrict__ Y) {
  __shared__ _Float16 As[32 * K] __attribute__((aligned(32)));
  __shared__ int sidx[64];
  const int e0 = blockIdx.x * 32;
  if (threadIdx.x < 32)      sidx[threadIdx.x] = dst[e0 + threadIdx.x];
  else if (threadIdx.x < 64) sidx[threadIdx.x] = src[e0 + threadIdx.x - 32];
  __syncthreads();
  if constexpr (CIN % 8 == 0) {                 // K == 2*CIN: fully vectorized gather
    for (int g = threadIdx.x; g < 4 * K; g += 256) {
      int r = g / (K / 8), kb = g % (K / 8);
      int k8 = kb * 8;
      v8h v;
      if (k8 < CIN) {
        v = *(const v8h*)&x[(size_t)sidx[r] * ldx + k8];
      } else {
        int cc = k8 - CIN;
        v8h xd = *(const v8h*)&x[(size_t)sidx[r]      * ldx + cc];
        v8h xs = *(const v8h*)&x[(size_t)sidx[32 + r] * ldx + cc];
        v = xs - xd;                            // packed f16 subtract
      }
      int c = k8 >> 5, kk = k8 & 31;
      int sub = r >> 4, lane16 = ((kk >> 3) & 1) * 16 + (r & 15);
      int jb = (kk & 16) >> 1;
      *(v8h*)&As[sub * 16 * K + (c * 32 + lane16) * 16 + jb] = v;
    }
  } else {                                      // gcu1: cin=3, Kpad=32, scalar + zero pad
    for (int i = threadIdx.x; i < 32 * K; i += 256) {
      int r = i / K, k = i % K;
      _Float16 v = (_Float16)0.0f;
      if (k < 2 * CIN) {
        int d = sidx[r];
        if (k < CIN) {
          v = x[(size_t)d * ldx + k];
        } else {
          int s = sidx[32 + r], cc = k - CIN;
          v = (_Float16)((float)x[(size_t)s * ldx + cc] - (float)x[(size_t)d * ldx + cc]);
        }
      }
      int c = k >> 5, kk = k & 31;
      int sub = r >> 4, lane16 = ((kk >> 3) & 1) * 16 + (r & 15);
      int j = (kk & 7) | ((kk & 16) >> 1);
      As[sub * 16 * K + (c * 32 + lane16) * 16 + j] = v;
    }
  }
  __syncthreads();
  wmma_tile_compute2<K, C>(As, Wp, bias, Y, e0);
}

// ---- deterministic two-stage BatchNorm stats ----
__global__ __launch_bounds__(256)
void bn_part(const float* __restrict__ Y, int R, int C, float* __restrict__ part) {
  const int p = blockIdx.x;
  const int rows = R / NPART;              // R in {393216, 65536}: divisible by 512
  const int r0 = p * rows;
  const int c = threadIdx.x;
  if (c >= C) return;
  float s = 0.0f, q = 0.0f;
  for (int r = r0; r < r0 + rows; ++r) {   // coalesced: thread == column
    float v = Y[(size_t)r * C + c];
    s += v; q += v * v;
  }
  part[(size_t)p * 2 * C + c]     = s;
  part[(size_t)p * 2 * C + C + c] = q;
}

__global__ __launch_bounds__(256)
void bn_finish(const float* __restrict__ part, int C, int R, float* __restrict__ musd) {
  __shared__ float ss[256], sq[256];
  const int c = blockIdx.x;
  float s = 0.0f, q = 0.0f;
  for (int p = threadIdx.x; p < NPART; p += 256) {
    s += part[(size_t)p * 2 * C + c];
    q += part[(size_t)p * 2 * C + C + c];
  }
  ss[threadIdx.x] = s; sq[threadIdx.x] = q;
  __syncthreads();
  for (int o = 128; o > 0; o >>= 1) {
    if (threadIdx.x < o) { ss[threadIdx.x] += ss[threadIdx.x + o];
                           sq[threadIdx.x] += sq[threadIdx.x + o]; }
    __syncthreads();
  }
  if (threadIdx.x == 0) {
    float mu  = ss[0] / (float)R;
    float var = sq[0] / (float)R - mu * mu;
    musd[c * 2]     = mu;
    musd[c * 2 + 1] = rsqrtf(var + BN_EPS);
  }
}

// ---- BN normalize (gamma/beta) -> f16, output row stride for concat slices ----
__global__ void bn_norm_f16(const float* __restrict__ Y, const float* __restrict__ musd,
                            const float* __restrict__ gamma, const float* __restrict__ beta,
                            _Float16* __restrict__ out, int R, int C, int ldo) {
  size_t i = (size_t)blockIdx.x * blockDim.x + threadIdx.x;
  if (i >= (size_t)R * C) return;
  int r = (int)(i / C), c = (int)(i % C);
  float v = (Y[i] - musd[c * 2]) * musd[c * 2 + 1] * gamma[c] + beta[c];
  out[(size_t)r * ldo + c] = (_Float16)v;
}

// ---- segment max via encoded atomicMax (deterministic) ----
__global__ void segmax_scatter(const _Float16* __restrict__ M, const int* __restrict__ seg,
                               unsigned* __restrict__ agg, int Rows, int C) {
  size_t i = (size_t)blockIdx.x * blockDim.x + threadIdx.x;
  if (i >= (size_t)Rows * C) return;
  int r = (int)(i / C), c = (int)(i % C);
  atomicMax(&agg[(size_t)seg[r] * C + c], enc_f32((float)M[i]));
}

// ---- decode two aggs (-inf -> 0) and concat into [N, 2C] f16 ----
__global__ void decode_concat2(const unsigned* __restrict__ a, const unsigned* __restrict__ b,
                               _Float16* __restrict__ xcat, int Nn, int C) {
  size_t i = (size_t)blockIdx.x * blockDim.x + threadIdx.x;
  if (i >= (size_t)Nn * C) return;
  int r = (int)(i / C), c = (int)(i % C);
  float va = (a[i] == ENC_NEGINF) ? 0.0f : dec_f32(a[i]);
  float vb = (b[i] == ENC_NEGINF) ? 0.0f : dec_f32(b[i]);
  xcat[(size_t)r * (2 * C) + c]     = (_Float16)va;
  xcat[(size_t)r * (2 * C) + C + c] = (_Float16)vb;
}

__global__ void decode_out(const unsigned* __restrict__ e, float* __restrict__ o, size_t n) {
  size_t i = (size_t)blockIdx.x * blockDim.x + threadIdx.x;
  if (i < n) o[i] = (e[i] == ENC_NEGINF) ? 0.0f : dec_f32(e[i]);
}

// ====================== host orchestration ======================
extern "C" void kernel_launch(void* const* d_in, const int* in_sizes, int n_in,
                              void* d_out, int out_size, void* d_ws, size_t ws_size,
                              hipStream_t stream) {
  (void)in_sizes; (void)n_in; (void)out_size; (void)ws_size;
  const float* pos    = (const float*)d_in[0];
  const int*   tpl_ei = (const int*)  d_in[1];   // [2,E]: row0=src, row1=dst
  const int*   geo_ei = (const int*)  d_in[2];
  const int*   batch  = (const int*)  d_in[3];

  char* ws = (char*)d_ws;
  size_t off = 0;
  auto alloc = [&](size_t bytes) -> char* {
    off = (off + 255) & ~(size_t)255;
    char* p = ws + off;
    off += bytes;
    return p;
  };

  _Float16* H    = (_Float16*)alloc((size_t)NEDGES * 128 * 2);
  _Float16* Mb   = (_Float16*)alloc((size_t)NEDGES * 128 * 2);
  float*    Yb   = (float*)   alloc((size_t)NEDGES * 128 * 4);
  unsigned* aggT = (unsigned*)alloc((size_t)NNODES * 128 * 4);
  unsigned* aggG = (unsigned*)alloc((size_t)NNODES * 128 * 4);
  _Float16* xcat = (_Float16*)alloc((size_t)NNODES * 256 * 2);
  _Float16* xg   = (_Float16*)alloc((size_t)NNODES * 448 * 2);  // [x1|x2|x3] columns
  _Float16* p16  = (_Float16*)alloc((size_t)NNODES * 3 * 2);
  _Float16* x4h  = (_Float16*)alloc((size_t)NNODES * 128 * 2);
  unsigned* oenc = (unsigned*)alloc((size_t)NGRAPHS * 128 * 4);
  float*    musd = (float*)   alloc((size_t)2 * 512 * 4);
  float*    part = (float*)   alloc((size_t)NPART * 2 * 256 * 4);

  // ---- params flatten order (jax pytree, sorted dict keys):
  // per gcu: geo[L0(W,b,bt,g),L1], mlp[L0], tpl[L0,L1]; then glb.
  struct WRec { int idx; int K; int Kpad; int C; _Float16* wh; };
  WRec wr[16];
  int nw = 0;
  const int cins[3] = {3, 64, 128};
  const int hids[3] = {32, 64, 128};
  for (int g = 0; g < 3; ++g) {
    int b = 4 + 20 * g;
    int K0 = 2 * cins[g], Kp0 = (K0 + 31) & ~31, hid = hids[g], cout = 2 * hid;
    wr[nw++] = { b + 0,  K0,  Kp0,  hid,  nullptr };   // geo L0
    wr[nw++] = { b + 4,  hid, hid,  hid,  nullptr };   // geo L1
    wr[nw++] = { b + 8,  cout,cout, cout, nullptr };   // mlp
    wr[nw++] = { b + 12, K0,  Kp0,  hid,  nullptr };   // tpl L0
    wr[nw++] = { b + 16, hid, hid,  hid,  nullptr };   // tpl L1
  }
  wr[nw++] = { 64, 448, 448, 128, nullptr };           // glb
  for (int i = 0; i < nw; ++i) {
    size_t tot = (size_t)wr[i].Kpad * wr[i].C;
    wr[i].wh = (_Float16*)alloc(tot * 2);
    conv_w_pack<<<(unsigned)((tot + 255) / 256), 256, 0, stream>>>(
        (const float*)d_in[wr[i].idx], wr[i].wh, wr[i].K, wr[i].C, wr[i].Kpad);
  }

  auto gemm = [&](const _Float16* A, const WRec& w, float* Y, int R) {
    dim3 grid(R / 32, (w.C / 16 + 7) / 8);
    const float* bias = (const float*)d_in[w.idx + 1];
    if      (w.Kpad == 32  && w.C == 32)
      wmma_gemm_t<32, 32>  <<<grid, 256, 0, stream>>>(A, w.wh, bias, Y);
    else if (w.Kpad == 64  && w.C == 64)
      wmma_gemm_t<64, 64>  <<<grid, 256, 0, stream>>>(A, w.wh, bias, Y);
    else if (w.Kpad == 128 && w.C == 128)
      wmma_gemm_t<128, 128><<<grid, 256, 0, stream>>>(A, w.wh, bias, Y);
    else if (w.Kpad == 256 && w.C == 256)
      wmma_gemm_t<256, 256><<<grid, 256, 0, stream>>>(A, w.wh, bias, Y);
    else if (w.Kpad == 448 && w.C == 128)
      wmma_gemm_t<448, 128><<<grid, 256, 0, stream>>>(A, w.wh, bias, Y);
  };
  auto edgegemm0 = [&](const _Float16* x, int ldx, const int* ei, const WRec& w, float* Y) {
    dim3 grid(NEDGES / 32, 1);       // C <= 128 -> <= 8 col tiles per block
    const float* bias = (const float*)d_in[w.idx + 1];
    if      (w.Kpad == 32)
      wmma_edge_gemm_t<32, 32, 3>    <<<grid, 256, 0, stream>>>(x, ldx, ei, ei + NEDGES,
                                                                w.wh, bias, Y);
    else if (w.Kpad == 128)
      wmma_edge_gemm_t<128, 64, 64>  <<<grid, 256, 0, stream>>>(x, ldx, ei, ei + NEDGES,
                                                                w.wh, bias, Y);
    else if (w.Kpad == 256)
      wmma_edge_gemm_t<256, 128, 128><<<grid, 256, 0, stream>>>(x, ldx, ei, ei + NEDGES,
                                                                w.wh, bias, Y);
  };
  auto bn = [&](float* Y, const WRec& w, int R, _Float16* out, int ldo) {
    bn_part  <<<NPART, 256, 0, stream>>>(Y, R, w.C, part);
    bn_finish<<<w.C,   256, 0, stream>>>(part, w.C, R, musd);
    size_t tot = (size_t)R * w.C;
    bn_norm_f16<<<(unsigned)((tot + 255) / 256), 256, 0, stream>>>(
        Y, musd, (const float*)d_in[w.idx + 3], (const float*)d_in[w.idx + 2],
        out, R, w.C, ldo);
  };
  auto edgeconv = [&](const _Float16* x, int ldx, const int* ei,
                      const WRec& w0, const WRec& w1, unsigned* agg) {
    int hid = w0.C;
    edgegemm0(x, ldx, ei, w0, Yb);            // fused gather + GEMM
    bn(Yb, w0, NEDGES, H, hid);
    gemm(H, w1, Yb, NEDGES);
    bn(Yb, w1, NEDGES, Mb, hid);
    size_t na = (size_t)NNODES * hid;
    fill_u32<<<(unsigned)((na + 255) / 256), 256, 0, stream>>>(agg, na, ENC_NEGINF);
    size_t te = (size_t)NEDGES * hid;
    segmax_scatter<<<(unsigned)((te + 255) / 256), 256, 0, stream>>>(
        Mb, ei + NEDGES, agg, NEDGES, hid);
  };

  // pos -> f16
  {
    size_t n = (size_t)NNODES * 3;
    conv_f32_f16<<<(unsigned)((n + 255) / 256), 256, 0, stream>>>(pos, p16, n);
  }

  const _Float16* xin = p16;
  int ldx = 3;
  const int coloff[3] = {0, 64, 192};     // x1:[0,64) x2:[64,192) x3:[192,448) in xg
  for (int g = 0; g < 3; ++g) {
    int hid = hids[g];
    edgeconv(xin, ldx, tpl_ei, wr[5 * g + 3], wr[5 * g + 4], aggT);
    edgeconv(xin, ldx, geo_ei, wr[5 * g + 0], wr[5 * g + 1], aggG);
    size_t tn = (size_t)NNODES * hid;
    decode_concat2<<<(unsigned)((tn + 255) / 256), 256, 0, stream>>>(
        aggT, aggG, xcat, NNODES, hid);
    gemm(xcat, wr[5 * g + 2], Yb, NNODES);
    bn(Yb, wr[5 * g + 2], NNODES, xg + coloff[g], 448);
    xin = xg + coloff[g];
    ldx = 448;
  }

  // glb MLP on [N,448], then per-graph segment max
  gemm(xg, wr[15], Yb, NNODES);
  bn(Yb, wr[15], NNODES, x4h, 128);
  size_t no = (size_t)NGRAPHS * 128;
  fill_u32<<<(unsigned)((no + 255) / 256), 256, 0, stream>>>(oenc, no, ENC_NEGINF);
  size_t tn = (size_t)NNODES * 128;
  segmax_scatter<<<(unsigned)((tn + 255) / 256), 256, 0, stream>>>(
      x4h, batch, oenc, NNODES, 128);
  decode_out<<<(unsigned)((no + 255) / 256), 256, 0, stream>>>(oenc, (float*)d_out, no);
}
